// PILES_61864708932006
// MI455X (gfx1250) — matrix operand order repaired and checked by reading, not compiled, not run on hardware
//
#include <hip/hip_runtime.h>
#include <hip/hip_bf16.h>
#include <stdint.h>

// Problem constants (match reference)
#define B_   32
#define S_   512
#define D_   512
#define HD_  512
#define M_   1000
#define NH_  8
#define DFF_ 2048
#define DH_  64
#define BS_  (B_ * S_)

typedef _Float16 half8  __attribute__((ext_vector_type(8)));
typedef _Float16 half16 __attribute__((ext_vector_type(16)));
typedef float    v8f    __attribute__((ext_vector_type(8)));
typedef unsigned int u32x4 __attribute__((ext_vector_type(4)));
typedef int      i32x4 __attribute__((ext_vector_type(4)));
typedef int      i32x8 __attribute__((ext_vector_type(8)));

static __device__ __forceinline__ half16 cat16(half8 lo, half8 hi) {
    return __builtin_shufflevector(lo, hi, 0,1,2,3,4,5,6,7,8,9,10,11,12,13,14,15);
}
static __device__ __forceinline__ v8f wmma_f16(half16 a, half16 b, v8f c) {
    // v_wmma_f32_16x16x32_f16: D = A(16x32) * B(32x16) + C(16x16 f32)
    return __builtin_amdgcn_wmma_f32_16x16x32_f16(false, a, false, b, (short)0, c, false, false);
}

// ---------------------------------------------------------------------------
// TDM: load a 2-D tile (tile_dim0=32 elems of 2B, tile_dim1=128 rows) from a
// row-major f16 tensor with row pitch `strideElems` into LDS at byte offset
// `ldsOff`, padding +16B after each 64B row -> LDS pitch 80B (40 halves).
// rem0/rem1 = remaining tensor extent from the tile origin (OOB reads -> 0).
// ---------------------------------------------------------------------------
static __device__ __forceinline__ void tdm_load_a_tile(
    const void* gsrc, unsigned ldsOff, int rem0, int rem1, int strideElems)
{
    unsigned long long ga = (unsigned long long)(uintptr_t)gsrc;
    u32x4 g0;
    g0[0] = 1u;                                        // count=1 (valid D#)
    g0[1] = ldsOff;                                    // lds_addr (bytes)
    g0[2] = (unsigned)(ga & 0xffffffffull);            // global_addr lo
    g0[3] = ((unsigned)(ga >> 32) & 0x01ffffffu)       // global_addr hi
          | 0x80000000u;                               // type=2 ("image")
    i32x8 g1;
    g1[0] = (int)((1u << 16)                           // data_size = 2B
                | (1u << 20)                           // pad_enable
                | (3u << 22)                           // pad_interval: 16 DW = 64B
                | (3u << 25));                         // pad_amount:   4 DW = 16B
    g1[1] = (int)(((unsigned)rem0 & 0xffffu) << 16);                       // dim0 lo16
    g1[2] = (int)(((unsigned)rem0 >> 16) | (((unsigned)rem1 & 0xffffu) << 16));
    g1[3] = (int)(((unsigned)rem1 >> 16) | (32u << 16));                   // tile_dim0=32
    g1[4] = 128;                                       // tile_dim1=128, tile_dim2=0
    g1[5] = strideElems;                               // tensor_dim0_stride lo32
    g1[6] = 0;
    g1[7] = 0;
    i32x4 gz = {0, 0, 0, 0};
#if __has_include(<hip/amd_detail/amd_gfx1250_TDM.h>)
    i32x8 gz8 = {0, 0, 0, 0, 0, 0, 0, 0};
    __builtin_amdgcn_tensor_load_to_lds(g0, g1, gz, gz, gz8, 0);   // clang-23 form
#else
    __builtin_amdgcn_tensor_load_to_lds(g0, g1, gz, gz, 0);        // ROCm 7.2 form
#endif
}

// ---------------------------------------------------------------------------
// f32 -> f16 weight conversion
// ---------------------------------------------------------------------------
__global__ void cvt_f32_to_f16(const float* __restrict__ src, _Float16* __restrict__ dst, int n) {
    int i = blockIdx.x * blockDim.x + threadIdx.x;
    if (i < n) dst[i] = (_Float16)src[i];
}

// ---------------------------------------------------------------------------
// Embedding: e = 0.5*W_m[:,mk] + 0.5*(W_t*t + b_t); zero + mask for padding
// ---------------------------------------------------------------------------
__global__ void embed_kernel(const float* __restrict__ x, const float* __restrict__ Wm,
                             const float* __restrict__ Wt, const float* __restrict__ bt,
                             _Float16* __restrict__ e16, float* __restrict__ mask) {
    int bs = blockIdx.x;                        // 0..BS-1
    float t  = x[(size_t)bs * 2 + 0];
    float mf = x[(size_t)bs * 2 + 1];
    int   mk = (int)fminf(fmaxf(mf, 0.0f), (float)(M_ - 1));
    bool  valid = (t >= 0.0f);
    if (threadIdx.x == 0) mask[bs] = valid ? 1.0f : 0.0f;
    for (int d = threadIdx.x; d < D_; d += blockDim.x) {
        float v = 0.0f;
        if (valid) v = 0.5f * Wm[(size_t)d * M_ + mk] + 0.5f * (Wt[d] * t + bt[d]);
        e16[(size_t)bs * D_ + d] = (_Float16)v;
    }
}

// ---------------------------------------------------------------------------
// WMMA GEMM: out[Mr,N] = A[Mr,K](f16) @ W[K,N](f16) (+bias)(relu)(+residual)
// Block = 256 threads = 8 waves (4x2), block tile 128x64, wave tile 32x32.
// A tile staged by the Tensor Data Mover (wave 0 issues, TENSORcnt waits),
// overlapped with the W-tile transpose staging done by all threads.
// LDS rows padded to 40 halves (80B) -> 16B-aligned b128 frags, no bank conflict.
// ---------------------------------------------------------------------------
__global__ __launch_bounds__(256) void gemm_wmma(
    const _Float16* __restrict__ A, const _Float16* __restrict__ W,
    float* __restrict__ out32, _Float16* __restrict__ out16,
    const float* __restrict__ bias,
    const _Float16* __restrict__ res16, const float* __restrict__ res32,
    int Mr, int N, int K, int ldc, int relu)
{
    __shared__ __align__(16) _Float16 ldsA[128 * 40];   // 128 rows x 32 K (padded)
    __shared__ __align__(16) _Float16 ldsW[64 * 40];    // 64 cols(n) x 32 K, transposed

    const int tid  = threadIdx.x;
    const int lane = tid & 31, l15 = lane & 15, lh = lane >> 4;
    const int wid  = tid >> 5;
    const int wm   = wid & 3, wn = wid >> 2;            // 4x2 wave grid
    const int n0   = blockIdx.x * 64;
    const long mb0 = (long)blockIdx.y * 128;

    // wave-uniform "is wave 0" predicate (scalar branch around TDM ops)
    const bool lead = (__builtin_amdgcn_readfirstlane((int)(tid >> 5)) == 0);
    const unsigned ldsA_off = (unsigned)(uintptr_t)(void*)ldsA;  // LDS byte offset
    const int remRows = (int)((long)Mr - mb0);          // rows left from tile origin

    const v8f vzero = {0.f,0.f,0.f,0.f,0.f,0.f,0.f,0.f};
    v8f acc[2][2];
    acc[0][0] = vzero; acc[0][1] = vzero; acc[1][0] = vzero; acc[1][1] = vzero;

    const int nk = K >> 5;
    for (int kb = 0; kb < nk; ++kb) {
        // ---- issue TDM for the A tile (128 x 32 halves, OOB rows -> 0) ----
        if (lead)
            tdm_load_a_tile(A + mb0 * (size_t)K + (size_t)kb * 32,
                            ldsA_off, K - kb * 32, remRows, K);
        // ---- stage W tile transposed: ldsW[n][k] (overlaps the DMA) ----
        {
            int i = tid * 8;
            int k = i >> 6, n = i & 63;                 // 64 n per k-row
            half8 v = *(const half8*)(W + (size_t)(kb * 32 + k) * N + n0 + n);
            if (kb + 1 < nk)
                __builtin_prefetch(W + (size_t)((kb + 1) * 32 + k) * N + n0 + n, 0, 1);
            #pragma unroll
            for (int j = 0; j < 8; ++j) ldsW[(n + j) * 40 + k] = v[j];
        }
        if (lead) __builtin_amdgcn_s_wait_tensorcnt(0);
        __syncthreads();

        half16 af[2], bf[2];
        #pragma unroll
        for (int mt = 0; mt < 2; ++mt) {
            // A frag: lane holds row M=l15, K = lh*8..+7 and 16+lh*8..+7
            const _Float16* p = ldsA + (wm * 32 + mt * 16 + l15) * 40 + lh * 8;
            af[mt] = cat16(*(const half8*)p, *(const half8*)(p + 16));
        }
        #pragma unroll
        for (int nt = 0; nt < 2; ++nt) {
            // B frag: lane holds col N=l15, K = lh*8..+7 and 16+lh*8..+7
            const _Float16* p = ldsW + (wn * 32 + nt * 16 + l15) * 40 + lh * 8;
            bf[nt] = cat16(*(const half8*)p, *(const half8*)(p + 16));
        }
        #pragma unroll
        for (int mt = 0; mt < 2; ++mt)
            #pragma unroll
            for (int nt = 0; nt < 2; ++nt)
                acc[mt][nt] = wmma_f16(af[mt], bf[nt], acc[mt][nt]);
        __syncthreads();   // all reads done before next iteration's TDM/stores
    }

    // ---- epilogue: C layout lane l -> (row = r + 8*lh, col = l15) ----
    #pragma unroll
    for (int mt = 0; mt < 2; ++mt) {
        #pragma unroll
        for (int nt = 0; nt < 2; ++nt) {
            int col = n0 + wn * 32 + nt * 16 + l15;
            float bcol = bias ? bias[col] : 0.0f;
            #pragma unroll
            for (int r = 0; r < 8; ++r) {
                long row = mb0 + wm * 32 + mt * 16 + r + 8 * lh;
                if (row < (long)Mr) {
                    float v = acc[mt][nt][r] + bcol;
                    if (relu) v = fmaxf(v, 0.0f);
                    size_t idx = (size_t)row * ldc + col;
                    if (res16) v += (float)res16[idx];
                    if (res32) v += res32[idx];
                    if (out32) out32[idx] = v;
                    if (out16) out16[idx] = (_Float16)v;
                }
            }
        }
    }
}

// ---------------------------------------------------------------------------
// Attention: one wave per (b, head, 16-query tile).
// Phase1: QK^T via WMMA (B-frags load contiguous from row-major K).
// Phase2: masked softmax rows in LDS (shfl_xor across half-waves).
// Phase3: P@V via WMMA with V staged transposed in LDS.
// Single-wave workgroup -> LDS ops are in-order within the wave (CDNA5).
// ---------------------------------------------------------------------------
__global__ __launch_bounds__(32) void attention_kernel(
    const _Float16* __restrict__ q16, const _Float16* __restrict__ k16,
    const _Float16* __restrict__ v16, const float* __restrict__ mask,
    _Float16* __restrict__ attn16)
{
    __shared__ float sS[16 * 512];                      // scores, f32
    __shared__ __align__(16) _Float16 sP[16 * 512];     // exp probs, f16
    __shared__ __align__(16) _Float16 sVt[64 * 40];     // V tile transposed [dh][key]
    __shared__ float sInv[16];

    const int lane = threadIdx.x, l15 = lane & 15, lh = lane >> 4;
    const int bid = blockIdx.x;
    const int qt  = bid & 31;                           // S/16 = 32 query tiles
    const int bh  = bid >> 5;
    const int h   = bh & (NH_ - 1);
    const int b   = bh >> 3;
    const int q0  = qt * 16;
    const size_t hoff = (size_t)h * DH_;
    const v8f vzero = {0.f,0.f,0.f,0.f,0.f,0.f,0.f,0.f};

    const _Float16* Qrow = q16 + ((size_t)(b * S_ + q0 + l15)) * D_ + hoff;

    // ---- phase 1: scores = Q K^T / sqrt(DH), masked ----
    for (int nt = 0; nt < 32; ++nt) {
        int key = nt * 16 + l15;
        const _Float16* Krow = k16 + ((size_t)(b * S_ + key)) * D_ + hoff;
        v8f acc = vzero;
        #pragma unroll
        for (int ks = 0; ks < 2; ++ks) {                // DH=64 -> 2 k-steps of 32
            half16 a  = cat16(*(const half8*)(Qrow + ks * 32 + lh * 8),
                              *(const half8*)(Qrow + ks * 32 + 16 + lh * 8));
            half16 bb = cat16(*(const half8*)(Krow + ks * 32 + lh * 8),
                              *(const half8*)(Krow + ks * 32 + 16 + lh * 8));
            acc = wmma_f16(a, bb, acc);
        }
        float mv = mask[b * S_ + key];
        #pragma unroll
        for (int r = 0; r < 8; ++r) {
            float s = acc[r] * 0.125f;                  // 1/sqrt(64)
            if (mv < 0.5f) s = -1e9f;
            sS[(r + 8 * lh) * 512 + key] = s;
        }
    }
    __syncthreads();

    // ---- phase 2: row softmax (unnormalized exp in sP, 1/sum in sInv) ----
    {
        int row = l15, start = lh * 256;
        const float* rp = sS + row * 512 + start;
        float mx = -3.0e38f;
        for (int j = 0; j < 256; ++j) mx = fmaxf(mx, rp[j]);
        mx = fmaxf(mx, __shfl_xor(mx, 16, 32));
        float sum = 0.0f;
        for (int j = 0; j < 256; ++j) {
            float e = __expf(rp[j] - mx);
            sP[row * 512 + start + j] = (_Float16)e;
            sum += e;
        }
        sum += __shfl_xor(sum, 16, 32);
        if (lh == 0) sInv[row] = 1.0f / sum;
    }
    __syncthreads();

    // ---- phase 3: O = P @ V ----
    v8f o[4];
    o[0] = vzero; o[1] = vzero; o[2] = vzero; o[3] = vzero;
    for (int kb = 0; kb < 16; ++kb) {                   // 512 keys / 32
        for (int i = lane; i < 32 * 64; i += 32) {      // stage V tile transposed
            int key = i >> 6, dh = i & 63;
            sVt[dh * 40 + key] = v16[((size_t)(b * S_ + kb * 32 + key)) * D_ + hoff + dh];
        }
        __syncthreads();
        half16 a;
        {
            const _Float16* p = sP + l15 * 512 + kb * 32 + lh * 8;
            a = cat16(*(const half8*)p, *(const half8*)(p + 16));
        }
        #pragma unroll
        for (int nt2 = 0; nt2 < 4; ++nt2) {
            const _Float16* p = sVt + (nt2 * 16 + l15) * 40 + lh * 8;
            half16 bb = cat16(*(const half8*)p, *(const half8*)(p + 16));
            o[nt2] = wmma_f16(a, bb, o[nt2]);
        }
        __syncthreads();
    }
    #pragma unroll
    for (int nt2 = 0; nt2 < 4; ++nt2) {
        int dh = nt2 * 16 + l15;
        #pragma unroll
        for (int r = 0; r < 8; ++r) {
            int row = r + 8 * lh;
            float val = o[nt2][r] * sInv[row];
            attn16[((size_t)(b * S_ + q0 + row)) * D_ + hoff + dh] = (_Float16)val;
        }
    }
}

// ---------------------------------------------------------------------------
// Masked mean pool over sequence: pooled16[b,d]
// ---------------------------------------------------------------------------
__global__ void pool_kernel(const float* __restrict__ h32, const float* __restrict__ mask,
                            _Float16* __restrict__ pooled16) {
    int b = blockIdx.x;
    float cnt = 0.0f;
    for (int s = 0; s < S_; ++s) cnt += mask[b * S_ + s];
    float inv = 1.0f / fmaxf(cnt, 1.0f);
    for (int d = threadIdx.x; d < D_; d += blockDim.x) {
        float sum = 0.0f;
        for (int s = 0; s < S_; ++s)
            sum += h32[((size_t)(b * S_ + s)) * D_ + d] * mask[b * S_ + s];
        pooled16[b * D_ + d] = (_Float16)(sum * inv);
    }
}

// ---------------------------------------------------------------------------
// Head: logits[b,m] = hcat[b,:] @ W_out[:,m] + b_out[m]  (tiny: plain VALU)
// ---------------------------------------------------------------------------
__global__ void head_kernel(const float* __restrict__ hcat, const float* __restrict__ Wout,
                            const float* __restrict__ bout, float* __restrict__ logits) {
    __shared__ float sh[2 * HD_];
    int b = blockIdx.x;
    for (int j = threadIdx.x; j < 2 * HD_; j += blockDim.x) sh[j] = hcat[b * 2 * HD_ + j];
    __syncthreads();
    for (int m = threadIdx.x; m < M_; m += blockDim.x) {
        float acc = bout[m];
        for (int j = 0; j < 2 * HD_; ++j) acc += sh[j] * Wout[(size_t)j * M_ + m];
        logits[b * M_ + m] = acc;
    }
}

// ---------------------------------------------------------------------------
// Row softmax -> lam, argmax -> mu (written as float after lam)
// ---------------------------------------------------------------------------
__global__ void softmax_argmax_kernel(const float* __restrict__ logits, float* __restrict__ out) {
    __shared__ float red[256];
    __shared__ int  redi[256];
    int b = blockIdx.x, tid = threadIdx.x;
    const float* row = logits + b * M_;
    float mx = -3.0e38f; int mi = 0;
    for (int m = tid; m < M_; m += 256) {
        float v = row[m];
        if (v > mx) { mx = v; mi = m; }
    }
    red[tid] = mx; redi[tid] = mi;
    __syncthreads();
    for (int sft = 128; sft > 0; sft >>= 1) {
        if (tid < sft) {
            float v2 = red[tid + sft]; int i2 = redi[tid + sft];
            if (v2 > red[tid] || (v2 == red[tid] && i2 < redi[tid])) {
                red[tid] = v2; redi[tid] = i2;
            }
        }
        __syncthreads();
    }
    float gmx = red[0]; int gidx = redi[0];
    __syncthreads();
    float sum = 0.0f;
    for (int m = tid; m < M_; m += 256) sum += __expf(row[m] - gmx);
    red[tid] = sum; __syncthreads();
    for (int sft = 128; sft > 0; sft >>= 1) {
        if (tid < sft) red[tid] += red[tid + sft];
        __syncthreads();
    }
    float inv = 1.0f / red[0];
    for (int m = tid; m < M_; m += 256)
        out[b * M_ + m] = __expf(row[m] - gmx) * inv;
    if (tid == 0) out[B_ * M_ + b] = (float)gidx;
}

// ---------------------------------------------------------------------------
extern "C" void kernel_launch(void* const* d_in, const int* in_sizes, int n_in,
                              void* d_out, int out_size, void* d_ws, size_t ws_size,
                              hipStream_t stream) {
    const float* x_H  = (const float*)d_in[0];
    const float* x_F  = (const float*)d_in[1];
    const float* W_m  = (const float*)d_in[2];
    const float* W_t  = (const float*)d_in[3];
    const float* b_t  = (const float*)d_in[4];
    const float* Wq   = (const float*)d_in[5];
    const float* Wk   = (const float*)d_in[6];
    const float* Wv   = (const float*)d_in[7];
    const float* Wo   = (const float*)d_in[8];
    const float* W1   = (const float*)d_in[9];
    const float* b1   = (const float*)d_in[10];
    const float* W2   = (const float*)d_in[11];
    const float* b2   = (const float*)d_in[12];
    const float* Wp   = (const float*)d_in[13];
    const float* bp   = (const float*)d_in[14];
    const float* Wout = (const float*)d_in[15];
    const float* bout = (const float*)d_in[16];
    float* out = (float*)d_out;

    char* ws = (char*)d_ws;
    size_t off = 0;
    auto alloc = [&](size_t bytes) -> char* {
        char* p = ws + off;
        off += (bytes + 255) & ~(size_t)255;
        return p;
    };
    _Float16* e16    = (_Float16*)alloc((size_t)BS_ * D_ * 2);
    _Float16* h16    = (_Float16*)alloc((size_t)BS_ * D_ * 2);
    float*    h32    = (float*)   alloc((size_t)BS_ * D_ * 4);
    _Float16* attn16 = (_Float16*)alloc((size_t)BS_ * D_ * 2);
    char*     un     = alloc((size_t)BS_ * DFF_ * 2);     // union: q/k/v then FFN mid
    _Float16* q16 = (_Float16*)un;
    _Float16* k16 = (_Float16*)(un + (size_t)BS_ * D_ * 2);
    _Float16* v16 = (_Float16*)(un + (size_t)BS_ * D_ * 4);
    _Float16* g16 = (_Float16*)un;                         // reuses q/k space post-attn
    float*    mask     = (float*)   alloc((size_t)B_ * S_ * 4);
    _Float16* pooled16 = (_Float16*)alloc((size_t)B_ * D_ * 2);
    float*    hcat     = (float*)   alloc((size_t)B_ * 2 * HD_ * 4);
    float*    logits   = (float*)   alloc((size_t)B_ * M_ * 4);
    _Float16* wq16 = (_Float16*)alloc((size_t)D_ * D_ * 2);
    _Float16* wk16 = (_Float16*)alloc((size_t)D_ * D_ * 2);
    _Float16* wv16 = (_Float16*)alloc((size_t)D_ * D_ * 2);
    _Float16* wo16 = (_Float16*)alloc((size_t)D_ * D_ * 2);
    _Float16* w116 = (_Float16*)alloc((size_t)D_ * DFF_ * 2);
    _Float16* w216 = (_Float16*)alloc((size_t)DFF_ * D_ * 2);
    _Float16* wp16 = (_Float16*)alloc((size_t)D_ * HD_ * 2);

    auto cvt = [&](const float* s, _Float16* d, int n) {
        cvt_f32_to_f16<<<(n + 255) / 256, 256, 0, stream>>>(s, d, n);
    };
    cvt(Wq, wq16, D_ * D_);
    cvt(Wk, wk16, D_ * D_);
    cvt(Wv, wv16, D_ * D_);
    cvt(Wo, wo16, D_ * D_);
    cvt(W1, w116, D_ * DFF_);
    cvt(W2, w216, DFF_ * D_);
    cvt(Wp, wp16, D_ * HD_);

    dim3 blk(256);
    dim3 gD(D_ / 64, (BS_ + 127) / 128);        // N=512 GEMMs over 16384 rows
    dim3 gF(DFF_ / 64, (BS_ + 127) / 128);      // N=2048
    dim3 gP(HD_ / 64, 1);                       // N=512 over 32 rows

    for (int enc = 0; enc < 2; ++enc) {
        const float* x = enc ? x_F : x_H;
        embed_kernel<<<BS_, 256, 0, stream>>>(x, W_m, W_t, b_t, e16, mask);

        gemm_wmma<<<gD, blk, 0, stream>>>(e16, wq16, nullptr, q16, nullptr, nullptr, nullptr,
                                          BS_, D_, D_, D_, 0);
        gemm_wmma<<<gD, blk, 0, stream>>>(e16, wk16, nullptr, k16, nullptr, nullptr, nullptr,
                                          BS_, D_, D_, D_, 0);
        gemm_wmma<<<gD, blk, 0, stream>>>(e16, wv16, nullptr, v16, nullptr, nullptr, nullptr,
                                          BS_, D_, D_, D_, 0);

        attention_kernel<<<B_ * NH_ * (S_ / 16), 32, 0, stream>>>(q16, k16, v16, mask, attn16);

        // h = e + attn @ Wo   (f32 stream + f16 shadow)
        gemm_wmma<<<gD, blk, 0, stream>>>(attn16, wo16, h32, h16, nullptr, e16, nullptr,
                                          BS_, D_, D_, D_, 0);
        // g = relu(h @ W1 + b1)
        gemm_wmma<<<gF, blk, 0, stream>>>(h16, w116, nullptr, g16, b1, nullptr, nullptr,
                                          BS_, DFF_, D_, DFF_, 1);
        // h = h + g @ W2 + b2  (in-place residual read-then-write per element)
        gemm_wmma<<<gD, blk, 0, stream>>>(g16, w216, h32, nullptr, b2, nullptr, h32,
                                          BS_, D_, DFF_, D_, 0);

        pool_kernel<<<B_, 256, 0, stream>>>(h32, mask, pooled16);
        // hcat[:, enc*512 : enc*512+512] = pooled @ Wp + bp   (ldc = 1024)
        gemm_wmma<<<gP, blk, 0, stream>>>(pooled16, wp16, hcat + enc * HD_, nullptr, bp,
                                          nullptr, nullptr, B_, HD_, D_, 2 * HD_, 0);
    }

    head_kernel<<<B_, 256, 0, stream>>>(hcat, Wout, bout, logits);
    softmax_argmax_kernel<<<B_, 256, 0, stream>>>(logits, out);
}
